// RegProxyAffinityHead_2147483648617
// MI455X (gfx1250) — matrix operand order repaired and checked by reading, not compiled, not run on hardware
//
#include <hip/hip_runtime.h>
#include <hip/hip_bf16.h>
#include <math.h>

// Problem constants (from reference)
#define B_  64
#define H_  24
#define W_  24
#define D_  1024
#define N_  (H_ * W_)   // 576
#define KC_ 256         // K_CLUSTERS; output is exactly C (top-256 centers)

typedef __attribute__((ext_vector_type(16))) __bf16 v16bf;
typedef __attribute__((ext_vector_type(8)))  float  v8f;
typedef __attribute__((ext_vector_type(4)))  unsigned int v4u;
typedef __attribute__((ext_vector_type(8)))  int  v8i;
typedef __attribute__((ext_vector_type(4)))  int  v4i;

union FragBf { v16bf v; unsigned int u[8]; };

__device__ __forceinline__ unsigned short f2bf(float f) {
  // round-to-nearest-even float -> bf16
  unsigned int u = __float_as_uint(f);
  unsigned int r = u + 0x7FFFu + ((u >> 16) & 1u);
  return (unsigned short)(r >> 16);
}

// ---------------------------------------------------------------------------
// Kernel 1: pack pointwise weights (9,D) -> bf16 matrix Wb[k=0..D-1][n=0..15]
// (rows 9..15 zero-padded). K-major so the WMMA B-fragment load is simple.
// ---------------------------------------------------------------------------
__global__ void prep_wb_kernel(const float* __restrict__ pw_w,
                               unsigned short* __restrict__ Wb) {
  int idx = blockIdx.x * blockDim.x + threadIdx.x;   // D_*16 total
  if (idx >= D_ * 16) return;
  int o = idx & 15;
  int d = idx >> 4;
  float v = (o < 9) ? pw_w[o * D_ + d] : 0.0f;
  Wb[d * 16 + o] = f2bf(v);
}

// ---------------------------------------------------------------------------
// Kernel 2: depthwise 3x3 conv (cross-correlation, zero pad 1) -> bf16 Xc(B,N,D)
// ---------------------------------------------------------------------------
__global__ void conv_dw_kernel(const float* __restrict__ tok2d,
                               const float* __restrict__ dw_w,
                               unsigned short* __restrict__ Xc) {
  int idx = blockIdx.x * blockDim.x + threadIdx.x;   // B_*N_*D_ total
  if (idx >= B_ * N_ * D_) return;
  int d = idx & (D_ - 1);
  int t = idx >> 10;            // D_ = 1024
  int p = t % N_;
  int b = t / N_;
  int y = p / W_, x = p % W_;
  float acc = 0.0f;
  #pragma unroll
  for (int ky = 0; ky < 3; ++ky) {
    int iy = y + ky - 1;
    if (iy < 0 || iy >= H_) continue;
    #pragma unroll
    for (int kx = 0; kx < 3; ++kx) {
      int ix = x + kx - 1;
      if (ix < 0 || ix >= W_) continue;
      size_t src = ((size_t)b * (N_ + 1) + 1 + (size_t)iy * W_ + ix) * D_ + d;
      acc += dw_w[d * 9 + ky * 3 + kx] * tok2d[src];
    }
  }
  Xc[idx] = f2bf(acc);
}

// ---------------------------------------------------------------------------
// Kernel 3: WMMA pointwise projection (16 tokens x 16 outs, K=1024 in 32-steps)
// + bias + softmax(9) + clipped-neighbor scatter-sum argmax -> parent.
// B-matrix (1024x16 bf16, 32KB) staged into LDS once per block with the
// Tensor Data Mover (TENSORcnt), then read via DS loads.
// One wave per 16-token tile; 4 tiles per 128-thread block.
// ---------------------------------------------------------------------------
#define TPB_TILES 4
__global__ void logits_parent_kernel(const unsigned short* __restrict__ Xc,
                                     const unsigned short* __restrict__ Wb,
                                     const float* __restrict__ pw_b,
                                     int* __restrict__ parent) {
  __shared__ unsigned short WbL[D_ * 16];            // 32KB weight tile
  __shared__ float ctile[TPB_TILES][16][16];         // 4KB C spill
  int wave = threadIdx.x >> 5;
  int lane = threadIdx.x & 31;
  int tile = blockIdx.x * TPB_TILES + wave;          // B_*N_/16 = 2304 tiles
  int b  = tile / (N_ / 16);
  int m0 = (tile % (N_ / 16)) * 16;

  // --- TDM: load the full 1024x16 bf16 weight tensor into LDS (wave 0) ---
  if (wave == 0) {
    unsigned long long ga = (unsigned long long)Wb;
    unsigned int lds = (unsigned int)(unsigned long long)(void*)WbL;
    v4u g0;
    g0.x = 1u;                                   // count=1, user descriptor
    g0.y = lds;                                  // lds_addr [63:32]
    g0.z = (unsigned int)(ga & 0xFFFFFFFFu);     // global_addr [95:64]
    g0.w = (unsigned int)((ga >> 32) & 0x01FFFFFFu) | (2u << 30); // addr hi | type=2
    v8i g1;
    g1[0] = (int)(1u << 16);        // workgroup_mask=0, data_size=1 (2 bytes)
    g1[1] = (int)(16u << 16);       // tensor_dim0 = 16 (low 16 bits @ 63:48)
    g1[2] = (int)(1024u << 16);     // tensor_dim0 hi=0 | tensor_dim1 = 1024 (lo16)
    g1[3] = (int)(16u << 16);       // tensor_dim1 hi=0 | tile_dim0 = 16
    g1[4] = 1024;                   // tile_dim1 = 1024 | tile_dim2 = 0
    g1[5] = 16;                     // tensor_dim0_stride = 16 (data_size units)
    g1[6] = 0;
    g1[7] = 0;
    v4i g2 = {0, 0, 0, 0};          // 2D tensor: groups 2/3 unused
    v4i g3 = {0, 0, 0, 0};
    v8i g4 = {0, 0, 0, 0, 0, 0, 0, 0};   // 6-arg toolchain: extra group, unused
    __builtin_amdgcn_tensor_load_to_lds(g0, g1, g2, g3, g4, 0);
    __builtin_amdgcn_s_wait_tensorcnt(0);
  }
  __syncthreads();                  // publish WbL to all 4 waves

  v8f c = {};
  FragBf a, bw;
  // A fragment (ISA 16-bit 16x32 layout): lane&15 = M row; lanes 0-15 K base 0,
  // lanes 16-31 K base 8; VGPR j holds K pair {off, off+1}.
  int M   = lane & 15;
  int akb = (lane < 16) ? 0 : 8;
  const unsigned short* arow = Xc + ((size_t)(b * N_ + m0 + M)) * D_;
  // B fragment: lane&15 = N col; same K layout as A (lanes 0-15 K base 0, lanes 16-31 K base 8).
  int n   = lane & 15;
  int bkb = (lane < 16) ? 0 : 8;

  for (int ks = 0; ks < D_; ks += 32) {
    #pragma unroll
    for (int j = 0; j < 8; ++j) {
      int ka = ks + akb + ((j < 4) ? 2 * j : 16 + 2 * (j - 4));
      a.u[j] = *(const unsigned int*)(arow + ka);    // two consecutive bf16
      int kb = ks + bkb + ((j < 4) ? 2 * j : 16 + 2 * (j - 4));
      unsigned int lo = WbL[kb * 16 + n];            // DS loads from staged tile
      unsigned int hi = WbL[(kb + 1) * 16 + n];
      bw.u[j] = lo | (hi << 16);
    }
    c = __builtin_amdgcn_wmma_f32_16x16x32_bf16(
            /*neg_a=*/false, a.v, /*neg_b=*/false, bw.v,
            /*c_mod=*/(short)0, c, /*reuse_a=*/false, /*reuse_b=*/false);
  }

  // Spill C to LDS (lane&15 = N col; VGPR r = row M, +8 for upper lanes).
  int mb = (lane < 16) ? 0 : 8;
  #pragma unroll
  for (int r = 0; r < 8; ++r) ctile[wave][mb + r][n] = c[r];
  __syncthreads();

  if (lane < 16) {
    int tok = m0 + lane;
    float l9[9];
    float mx = -1e30f;
    #pragma unroll
    for (int o = 0; o < 9; ++o) {
      float v = ctile[wave][lane][o] + pw_b[o];
      l9[o] = v;
      mx = fmaxf(mx, v);
    }
    float q[9];
    #pragma unroll
    for (int o = 0; o < 9; ++o) q[o] = __expf(l9[o] - mx);
    // softmax normalization is a positive per-token scale -> argmax-invariant;
    // skip the divide.
    int y = tok / W_, x = tok % W_;
    int nidx[9];
    #pragma unroll
    for (int k = 0; k < 9; ++k) {
      int ny = y + (k / 3) - 1;
      int nx = x + (k % 3) - 1;
      ny = ny < 0 ? 0 : (ny > H_ - 1 ? H_ - 1 : ny);
      nx = nx < 0 ? 0 : (nx > W_ - 1 ? W_ - 1 : nx);
      nidx[k] = ny * W_ + nx;
    }
    float bs = -1e30f;
    int   bn = 0x7fffffff;
    #pragma unroll
    for (int k = 0; k < 9; ++k) {
      int nk = nidx[k];
      float s = 0.0f;
      #pragma unroll
      for (int j = 0; j < 9; ++j) s += (nidx[j] == nk) ? q[j] : 0.0f;
      if (s > bs || (s == bs && nk < bn)) { bs = s; bn = nk; }
    }
    parent[b * N_ + tok] = bn;
  }
}

// ---------------------------------------------------------------------------
// Kernel 4: roots = parent applied 2^6 = 64 times (matches 6 doubling iters)
// ---------------------------------------------------------------------------
__global__ void find_roots_kernel(const int* __restrict__ parent,
                                  int* __restrict__ root) {
  int idx = blockIdx.x * blockDim.x + threadIdx.x;
  if (idx >= B_ * N_) return;
  int b = idx / N_;
  int r = idx % N_;
  const int* pb = parent + b * N_;
  #pragma unroll 8
  for (int it = 0; it < 64; ++it) r = pb[r];
  root[idx] = r;
}

// ---------------------------------------------------------------------------
// Kernel 5 (per batch, 576 threads): counts + rank-based top-256 selection.
// rank_i = #{j : counts_j > counts_i or (== and j<i)}; selected iff rank<256
// (== top_k tie semantics). slot = prefix count of selected (== sort asc).
// ---------------------------------------------------------------------------
__global__ void topk_counts_kernel(const int* __restrict__ root,
                                   int* __restrict__ root2slot,
                                   float* __restrict__ selCounts) {
  __shared__ int icnt[N_];
  __shared__ int sflag[N_];
  int b = blockIdx.x;
  int i = threadIdx.x;          // 0..575
  icnt[i] = 0;
  __syncthreads();
  atomicAdd(&icnt[root[b * N_ + i]], 1);
  __syncthreads();
  int my = icnt[i];
  int rank = 0;
  for (int j = 0; j < N_; ++j) {
    int cj = icnt[j];
    rank += (cj > my) || (cj == my && j < i);
  }
  int sel = (rank < KC_) ? 1 : 0;
  sflag[i] = sel;
  __syncthreads();
  int slot = 0;
  for (int j = 0; j < i; ++j) slot += sflag[j];
  if (sel) selCounts[b * KC_ + slot] = (float)my;
  root2slot[b * N_ + i] = sel ? slot : -1;
}

// ---------------------------------------------------------------------------
// Kernel 6 (per (batch,slot), 256 threads): gather member tokens, average,
// write final centers into d_out. Full coverage of d_out -> no pre-zeroing.
// ---------------------------------------------------------------------------
__global__ void gather_avg_kernel(const float* __restrict__ tok2d,
                                  const int* __restrict__ root,
                                  const int* __restrict__ root2slot,
                                  const float* __restrict__ selCounts,
                                  float* __restrict__ out) {
  __shared__ int member[N_];
  int b = blockIdx.x >> 8;
  int s = blockIdx.x & 255;
  int tid = threadIdx.x;
  for (int i = tid; i < N_; i += 256)
    member[i] = (root2slot[b * N_ + root[b * N_ + i]] == s);
  __syncthreads();
  const float* P = tok2d + ((size_t)b * (N_ + 1) + 1) * D_;
  float a0 = 0.f, a1 = 0.f, a2 = 0.f, a3 = 0.f;
  for (int i = 0; i < N_; ++i) {
    if (member[i]) {                      // uniform branch across the block
      const float* row = P + (size_t)i * D_;
      if (i + 1 < N_ && member[i + 1])    // hint: global_prefetch_b8
        __builtin_prefetch(P + (size_t)(i + 1) * D_ + tid, 0, 3);
      a0 += row[tid];
      a1 += row[tid + 256];
      a2 += row[tid + 512];
      a3 += row[tid + 768];
    }
  }
  float cnt = selCounts[b * KC_ + s];
  float inv = 1.0f / fmaxf(cnt, 1.0f);
  float* o = out + ((size_t)b * KC_ + s) * D_;
  o[tid]       = a0 * inv;
  o[tid + 256] = a1 * inv;
  o[tid + 512] = a2 * inv;
  o[tid + 768] = a3 * inv;
}

// ---------------------------------------------------------------------------
extern "C" void kernel_launch(void* const* d_in, const int* in_sizes, int n_in,
                              void* d_out, int out_size, void* d_ws, size_t ws_size,
                              hipStream_t stream) {
  (void)in_sizes; (void)n_in; (void)out_size; (void)ws_size;
  const float* tok2d = (const float*)d_in[0];   // (B, N+1, D)
  const float* dw_w  = (const float*)d_in[1];   // (D, 1, 3, 3)
  const float* pw_w  = (const float*)d_in[2];   // (9, D, 1, 1)
  const float* pw_b  = (const float*)d_in[3];   // (9,)
  // d_in[4..7] = q_w, q_b, v_w, v_b: dead code — reference's top_k(valid,256)
  // over a 256-length axis selects everything; sorted indices are identity,
  // so sel == C (the averaged top-256 cluster centers).
  float* out = (float*)d_out;                   // (B, 256, D) f32

  // Workspace carve-out (~76 MB)
  char* ws = (char*)d_ws;
  size_t off = 0;
  auto carve = [&](size_t bytes) -> char* {
    char* p = ws + off;
    off = (off + bytes + 255) & ~(size_t)255;
    return p;
  };
  unsigned short* Xc        = (unsigned short*)carve((size_t)B_ * N_ * D_ * 2);
  unsigned short* Wb        = (unsigned short*)carve((size_t)D_ * 16 * 2);
  int*            parent    = (int*)carve((size_t)B_ * N_ * 4);
  int*            root      = (int*)carve((size_t)B_ * N_ * 4);
  int*            root2slot = (int*)carve((size_t)B_ * N_ * 4);
  float*          selCounts = (float*)carve((size_t)B_ * KC_ * 4);

  // 1. pack pointwise weights to bf16 (16-wide, zero-padded)
  prep_wb_kernel<<<(D_ * 16 + 255) / 256, 256, 0, stream>>>(pw_w, Wb);
  // 2. depthwise conv -> bf16
  conv_dw_kernel<<<(B_ * N_ * D_ + 255) / 256, 256, 0, stream>>>(tok2d, dw_w, Xc);
  // 3. TDM-staged WMMA projection + softmax + argmax -> parent
  logits_parent_kernel<<<(B_ * N_ / 16) / TPB_TILES, 32 * TPB_TILES, 0, stream>>>(
      Xc, Wb, pw_b, parent);
  // 4. pointer jumping (64 steps)
  find_roots_kernel<<<(B_ * N_ + 255) / 256, 256, 0, stream>>>(parent, root);
  // 5. counts + top-256 selection per batch
  topk_counts_kernel<<<B_, N_, 0, stream>>>(root, root2slot, selCounts);
  // 6. gather + average -> output
  gather_avg_kernel<<<B_ * KC_, 256, 0, stream>>>(tok2d, root, root2slot,
                                                  selCounts, out);
}
